// MultiHeadAttention_28054726377981
// MI455X (gfx1250) — matrix operand order repaired and checked by reading, not compiled, not run on hardware
//
#include <hip/hip_runtime.h>
#include <hip/hip_bf16.h>
#include <stdint.h>

#define H_NUM 16
#define B_NUM 2
#define L_SEQ 2048
#define D_MOD 1024
#define D_HEAD 64

typedef __attribute__((ext_vector_type(16))) __bf16 v16bf;
typedef __attribute__((ext_vector_type(8)))  float  v8f;
typedef __attribute__((ext_vector_type(4)))  int    v4i_t;

union FragBF {
  uint32_t u[8];
  v16bf    v;
};

static __device__ __forceinline__ uint16_t f2bf(float f) {
  uint32_t u = __float_as_uint(f);
  u += 0x7FFFu + ((u >> 16) & 1u);
  return (uint16_t)(u >> 16);
}

static __device__ __forceinline__ v8f wmma_bf16(const FragBF& a, const FragBF& b, v8f c) {
  // D = A(16x32 bf16) * B(32x16 bf16) + C(16x16 f32)
  return __builtin_amdgcn_wmma_f32_16x16x32_bf16(false, a.v, false, b.v,
                                                 (short)0, c, false, false);
}

// 16-bit A-matrix 16x32 layout (contiguous k order, matches B): VGPR v of lane (g=lane/16) holds
// K = kb, kb+1 with kb = 2v + 16g
static __device__ __forceinline__ int a_kbase(int v, int g) {
  return 2 * v + 16 * g;
}

// --------------------------------------------------------------------------
// gfx1250 async global->LDS DMA (signature confirmed by compiler diagnostic:
// (int4 AS1*, int4 AS3*, imm, imm cpol)); fallback: register-staged b128.
// --------------------------------------------------------------------------
#if defined(__has_builtin)
#if __has_builtin(__builtin_amdgcn_global_load_async_to_lds_b128)
#define HAVE_ASYNC_LDS 1
#endif
#endif

#ifdef HAVE_ASYNC_LDS
#define GLOBAL_AS __attribute__((address_space(1)))
#define LDS_AS    __attribute__((address_space(3)))
static __device__ __forceinline__ void async_copy16(const uint16_t* g, uint16_t* l) {
  __builtin_amdgcn_global_load_async_to_lds_b128(
      (GLOBAL_AS v4i_t*)g, (LDS_AS v4i_t*)l, 0, 0);
}
static __device__ __forceinline__ void wait_async0() {
#if __has_builtin(__builtin_amdgcn_s_wait_asynccnt)
  __builtin_amdgcn_s_wait_asynccnt(0);
#else
  asm volatile("s_wait_asynccnt 0x0" ::: "memory");
#endif
}
#endif

// ---------------------------------------------------------------------------
// f32 -> bf16 conversion (2 elements / thread, packed u32 store)
// ---------------------------------------------------------------------------
__global__ void f32_to_bf16_kernel(const float* __restrict__ src,
                                   uint16_t* __restrict__ dst, int npairs) {
  int i = blockIdx.x * blockDim.x + threadIdx.x;
  if (i < npairs) {
    float2 f = ((const float2*)src)[i];
    uint32_t lo = f2bf(f.x);
    uint32_t hi = f2bf(f.y);
    ((uint32_t*)dst)[i] = lo | (hi << 16);
  }
}

// ---------------------------------------------------------------------------
// Weight convert + swizzle into WMMA B-fragment order.
// For K-tile kt (32 K-rows), element (k, n) of the 1024x1024 weight goes to
// u16 index  kt*32768 + n*32 + (k & 31)   i.e. layout [kt][n][kpair(16)][2].
// Each (kt, 64-column) B tile is then one contiguous 4 KB block, and LDS
// B-fragment reads become two ds_load_b128 per fragment (same as A).
// Exactly 2^19 u32 outputs -> grid covers it exactly.
// ---------------------------------------------------------------------------
__global__ __launch_bounds__(256)
void weight_swizzle_bf16_kernel(const float* __restrict__ W,
                                uint16_t* __restrict__ Wsw) {
  int i = blockIdx.x * blockDim.x + threadIdx.x;  // u32 element index
  int kt = i >> 14;           // K-tile (32 tiles of 32 rows)
  int kp = (i >> 10) & 15;    // K-pair within tile
  int n  = i & 1023;          // column (fastest -> coalesced reads)
  int k  = kt * 32 + kp * 2;
  float lo = W[(size_t)k * D_MOD + n];
  float hi = W[(size_t)(k + 1) * D_MOD + n];
  uint32_t p = (uint32_t)f2bf(lo) | ((uint32_t)f2bf(hi) << 16);
  ((uint32_t*)Wsw)[(size_t)kt * 16384 + (size_t)n * 16 + kp] = p;
}

// ---------------------------------------------------------------------------
// bf16 WMMA GEMM:  C(f32, Mx1024) = alpha * A(bf16 row-major, Mx1024)
//                                         @ B(bf16 swizzled, 1024x1024)
// Block tile 128x64, 8 waves (4 along M, 2 along N), each wave 32x32 (2x2 WMMA).
// Async-DMA double-buffered LDS; 3x b128 per thread per K-tile.
// ---------------------------------------------------------------------------
#define TBM 128
#define TBN 64
#define TBK 32
#define LDSA 40   // u16 row stride for As: 80 B rows (16 B multiple)
#define LDSB 40   // u16 stride per B column: 32 payload + 8 pad (80 B)

__global__ __launch_bounds__(256)
void gemm_bf16_wmma(const uint16_t* __restrict__ A,
                    const uint16_t* __restrict__ Bsw,
                    float* __restrict__ C, float alpha) {
  __shared__ __align__(16) uint16_t As[2][TBM * LDSA];
  __shared__ __align__(16) uint16_t Bs[2][TBN * LDSB];

  const int tid  = threadIdx.x;
  const int lane = tid & 31;
  const int wave = tid >> 5;
  const int wm   = wave & 3;   // M quadrant (32 rows)
  const int wn   = wave >> 2;  // N half (32 cols)
  const int l16  = lane & 15;
  const int g    = lane >> 4;

  const int rowBase = blockIdx.y * TBM;
  const int colBase = blockIdx.x * TBN;

  // Staging: A rows tid>>2 and +64 (4 b128/row); B column tid>>2 (4 b128/col).
  const int arow = tid >> 2, aq = tid & 3;
  const uint16_t* Aptr = A + (size_t)(rowBase + arow) * D_MOD + aq * 8;
  const uint16_t* Bptr = Bsw + (size_t)(colBase + arow) * 32 + aq * 8;
  const int asOff0 = arow * LDSA + aq * 8;
  const int asOff1 = (arow + 64) * LDSA + aq * 8;
  const int bsOff  = arow * LDSB + aq * 8;

  v8f acc[2][2];
  for (int mt = 0; mt < 2; ++mt)
    for (int nt = 0; nt < 2; ++nt)
      acc[mt][nt] = v8f{};

#ifdef HAVE_ASYNC_LDS
  // ---- async pipeline: DMA tile kt+1 into buf^1 while WMMA consumes buf ----
  async_copy16(Aptr, &As[0][asOff0]);
  async_copy16(Aptr + 64 * D_MOD, &As[0][asOff1]);
  async_copy16(Bptr, &Bs[0][bsOff]);
  wait_async0();
  __syncthreads();
#else
  uint4 ra0 = *(const uint4*)(Aptr);
  uint4 ra1 = *(const uint4*)(Aptr + 64 * D_MOD);
  uint4 rb  = *(const uint4*)(Bptr);
#endif

  for (int k0 = 0; k0 < D_MOD; k0 += TBK) {
    const int kt  = k0 >> 5;
    const int buf = kt & 1;

#ifdef HAVE_ASYNC_LDS
    if (k0 + TBK < D_MOD) {  // issue next tile's DMA before compute
      async_copy16(Aptr + k0 + TBK, &As[buf ^ 1][asOff0]);
      async_copy16(Aptr + k0 + TBK + 64 * D_MOD, &As[buf ^ 1][asOff1]);
      async_copy16(Bptr + (size_t)(kt + 1) * 32768, &Bs[buf ^ 1][bsOff]);
    }
#else
    *(uint4*)&As[buf][asOff0] = ra0;
    *(uint4*)&As[buf][asOff1] = ra1;
    *(uint4*)&Bs[buf][bsOff]  = rb;
    __syncthreads();
    if (k0 + TBK < D_MOD) {  // global loads for tile k+1 fly during compute
      ra0 = *(const uint4*)(Aptr + k0 + TBK);
      ra1 = *(const uint4*)(Aptr + k0 + TBK + 64 * D_MOD);
      rb  = *(const uint4*)(Bptr + (size_t)(kt + 1) * 32768);
    }
#endif

    FragBF fragA[2], fragB[2];
#pragma unroll
    for (int mt = 0; mt < 2; ++mt) {
      int row = wm * 32 + mt * 16 + l16;
#pragma unroll
      for (int v = 0; v < 8; ++v)
        fragA[mt].u[v] = *(const uint32_t*)&As[buf][row * LDSA + a_kbase(v, g)];
    }
#pragma unroll
    for (int nt = 0; nt < 2; ++nt) {
      int col = wn * 32 + nt * 16 + l16;
#pragma unroll
      for (int v = 0; v < 8; ++v)  // K pair (8g+v) of this column
        fragB[nt].u[v] = *(const uint32_t*)&Bs[buf][col * LDSB + 16 * g + 2 * v];
    }
#pragma unroll
    for (int mt = 0; mt < 2; ++mt)
#pragma unroll
      for (int nt = 0; nt < 2; ++nt)
        acc[mt][nt] = wmma_bf16(fragA[mt], fragB[nt], acc[mt][nt]);

#ifdef HAVE_ASYNC_LDS
    wait_async0();     // next tile landed in LDS
    __syncthreads();   // publish to all waves; also closes reads of buf
#endif
  }

#ifndef HAVE_ASYNC_LDS
  __syncthreads();
#endif

  // C/D layout: VGPR r, lane -> row = r + 8*(lane/16), col = lane%16
#pragma unroll
  for (int mt = 0; mt < 2; ++mt)
#pragma unroll
    for (int nt = 0; nt < 2; ++nt)
#pragma unroll
      for (int r = 0; r < 8; ++r) {
        int row = rowBase + wm * 32 + mt * 16 + r + 8 * g;
        int col = colBase + wn * 32 + nt * 16 + l16;
        C[(size_t)row * D_MOD + col] = acc[mt][nt][r] * alpha;
      }
}

// ---------------------------------------------------------------------------
// RoPE + head split. Reads fp32 Q/K/V projections [B*L, D], writes bf16:
//   Qh, Kh : [b][h][l][64]   (rope applied)
//   Vt     : [b][h][64][l]   (transposed so PV B-fragments load as u32 pairs)
// ---------------------------------------------------------------------------
__global__ __launch_bounds__(256)
void rope_split_kernel(const float* __restrict__ Qf,
                       const float* __restrict__ Kf,
                       const float* __restrict__ Vf,
                       uint16_t* __restrict__ Qh,
                       uint16_t* __restrict__ Kh,
                       uint16_t* __restrict__ Vt) {
  const int row = blockIdx.x;          // 0 .. B*L-1
  const int b = row / L_SEQ;
  const int l = row % L_SEQ;
  const float LOG2_10000 = 13.2877123795494f;

  for (int p = threadIdx.x; p < D_MOD / 2; p += blockDim.x) {
    int h = p >> 5;                    // 32 pairs per 64-dim head
    int i = p & 31;                    // pair index inside head
    float ang = (float)l * exp2f(-((float)(2 * i) / 64.0f) * LOG2_10000);
    float sn = sinf(ang);
    float cs = cosf(ang);

    size_t src = (size_t)row * D_MOD + 2 * p;
    float q1 = Qf[src], q2 = Qf[src + 1];
    float k1 = Kf[src], k2 = Kf[src + 1];
    float v1 = Vf[src], v2 = Vf[src + 1];

    size_t dst = (((size_t)(b * H_NUM + h)) * L_SEQ + l) * D_HEAD + 2 * i;
    Qh[dst]     = f2bf(q1 * cs - q2 * sn);
    Qh[dst + 1] = f2bf(q2 * cs + q1 * sn);
    Kh[dst]     = f2bf(k1 * cs - k2 * sn);
    Kh[dst + 1] = f2bf(k2 * cs + k1 * sn);

    size_t vdst = ((size_t)(b * H_NUM + h) * D_HEAD + 2 * i) * L_SEQ + l;
    Vt[vdst]         = f2bf(v1);
    Vt[vdst + L_SEQ] = f2bf(v2);
  }
}

// ---------------------------------------------------------------------------
// Flash-style causal attention, one wave per 16-row Q tile, 32 keys per step.
// Block = 4 waves (128 threads); grid = (L/64, H, B). Output Y bf16 [B*L, D].
// ---------------------------------------------------------------------------
__global__ __launch_bounds__(128)
void attention_wmma_kernel(const uint16_t* __restrict__ Qh,
                           const uint16_t* __restrict__ Kh,
                           const uint16_t* __restrict__ Vt,
                           uint16_t* __restrict__ Y) {
  __shared__ __align__(16) uint16_t Plds[4][16 * 32];  // per-wave P staging

  const int lane = threadIdx.x & 31;
  const int wave = threadIdx.x >> 5;
  const int l16  = lane & 15;
  const int g    = lane >> 4;

  const int h = blockIdx.y;
  const int b = blockIdx.z;
  const int qbase = (blockIdx.x * 4 + wave) * 16;
  const int bh = b * H_NUM + h;

  const uint16_t* Qp = Qh + (size_t)bh * L_SEQ * D_HEAD;
  const uint16_t* Kp = Kh + (size_t)bh * L_SEQ * D_HEAD;
  const uint16_t* Vp = Vt + (size_t)bh * D_HEAD * L_SEQ;

  // Q A-fragments for the two 32-wide K chunks of head_dim=64.
  FragBF fragQ[2];
#pragma unroll
  for (int c = 0; c < 2; ++c)
#pragma unroll
    for (int v = 0; v < 8; ++v)
      fragQ[c].u[v] =
          *(const uint32_t*)&Qp[(size_t)(qbase + l16) * D_HEAD + c * 32 + a_kbase(v, g)];

  float mrow[8], lrow[8];
  v8f oacc[4];
#pragma unroll
  for (int r = 0; r < 8; ++r) { mrow[r] = -3.0e38f; lrow[r] = 0.0f; }
#pragma unroll
  for (int nt = 0; nt < 4; ++nt) oacc[nt] = v8f{};

  const int numJ = (qbase + 47) / 32;  // key tiles of 32, causal bound
  for (int j = 0; j < numJ; ++j) {
    const int kb0 = j * 32;

    // S = Q K^T  (two 16x16 tiles)
    v8f s[2];
    s[0] = v8f{}; s[1] = v8f{};
#pragma unroll
    for (int st = 0; st < 2; ++st) {
      FragBF fk0, fk1;
      const int key = kb0 + st * 16 + l16;
#pragma unroll
      for (int v = 0; v < 8; ++v) {
        int kc = 16 * g + 2 * v;
        fk0.u[v] = *(const uint32_t*)&Kp[(size_t)key * D_HEAD + kc];
        fk1.u[v] = *(const uint32_t*)&Kp[(size_t)key * D_HEAD + 32 + kc];
      }
      s[st] = wmma_bf16(fragQ[0], fk0, s[st]);
      s[st] = wmma_bf16(fragQ[1], fk1, s[st]);
    }

    // Online softmax over this 32-key block (per-row state replicated
    // across each 16-lane half; reductions via shfl_xor within halves).
#pragma unroll
    for (int r = 0; r < 8; ++r) {
      const int qa = qbase + r + 8 * g;
      float s0 = s[0][r];
      float s1 = s[1][r];
      if (kb0 + l16 > qa)      s0 = -1.0e9f;   // causal mask
      if (kb0 + 16 + l16 > qa) s1 = -1.0e9f;

      float tmax = fmaxf(s0, s1);
#pragma unroll
      for (int m = 1; m < 16; m <<= 1) tmax = fmaxf(tmax, __shfl_xor(tmax, m, 32));
      const float mnew  = fmaxf(mrow[r], tmax);
      const float scale = __expf(mrow[r] - mnew);
      mrow[r] = mnew;

      const float p0 = __expf(s0 - mnew);
      const float p1 = __expf(s1 - mnew);
      float ps = p0 + p1;
#pragma unroll
      for (int m = 1; m < 16; m <<= 1) ps += __shfl_xor(ps, m, 32);
      lrow[r] = lrow[r] * scale + ps;
#pragma unroll
      for (int nt = 0; nt < 4; ++nt) oacc[nt][r] *= scale;

      const int prow = r + 8 * g;   // C-layout row -> LDS row
      Plds[wave][prow * 32 + l16]      = f2bf(p0);
      Plds[wave][prow * 32 + 16 + l16] = f2bf(p1);
    }

    // Reload P as an A-fragment (per-wave LDS is hardware-ordered in-wave).
    FragBF fragP;
#pragma unroll
    for (int v = 0; v < 8; ++v)
      fragP.u[v] = *(const uint32_t*)&Plds[wave][l16 * 32 + a_kbase(v, g)];

    // O += P @ V  (V transposed layout makes every pair a contiguous u32)
#pragma unroll
    for (int nt = 0; nt < 4; ++nt) {
      FragBF fv;
      const int col = nt * 16 + l16;
#pragma unroll
      for (int v = 0; v < 8; ++v) {
        int kc = 16 * g + 2 * v;
        fv.u[v] = *(const uint32_t*)&Vp[(size_t)col * L_SEQ + kb0 + kc];
      }
      oacc[nt] = wmma_bf16(fragP, fv, oacc[nt]);
    }
  }

  // Normalize and merge heads: Y[b*L + q][h*64 + col] (bf16 for final GEMM).
#pragma unroll
  for (int nt = 0; nt < 4; ++nt)
#pragma unroll
    for (int r = 0; r < 8; ++r) {
      const int qrow = qbase + r + 8 * g;
      const float val = oacc[nt][r] / lrow[r];
      Y[(size_t)(b * L_SEQ + qrow) * D_MOD + h * D_HEAD + nt * 16 + l16] = f2bf(val);
    }
}

// ---------------------------------------------------------------------------
// Host-side orchestration
// ---------------------------------------------------------------------------
extern "C" void kernel_launch(void* const* d_in, const int* in_sizes, int n_in,
                              void* d_out, int out_size, void* d_ws, size_t ws_size,
                              hipStream_t stream) {
  (void)in_sizes; (void)n_in; (void)out_size; (void)ws_size;
  const float* x  = (const float*)d_in[0];
  // d_in[1] is the additive causal mask; we fold causality in directly.
  const float* Wq = (const float*)d_in[2];
  const float* Wk = (const float*)d_in[3];
  const float* Wv = (const float*)d_in[4];
  const float* Wo = (const float*)d_in[5];
  float* out = (float*)d_out;

  const int Mrows = B_NUM * L_SEQ;          // 4096
  const int Dsq   = D_MOD * D_MOD;          // 1048576
  const int Xn    = Mrows * D_MOD;          // 4194304

  char* w = (char*)d_ws;
  size_t off = 0;
  auto carve = [&](size_t bytes) -> void* {
    void* p = w + off;
    off += (bytes + 255) & ~(size_t)255;
    return p;
  };

  uint16_t* Xb  = (uint16_t*)carve((size_t)Xn * 2);
  uint16_t* Wqb = (uint16_t*)carve((size_t)Dsq * 2);  // swizzled
  uint16_t* Wkb = (uint16_t*)carve((size_t)Dsq * 2);  // swizzled
  uint16_t* Wvb = (uint16_t*)carve((size_t)Dsq * 2);  // swizzled
  uint16_t* Wob = (uint16_t*)carve((size_t)Dsq * 2);  // swizzled
  float*    Qf  = (float*)carve((size_t)Xn * 4);
  float*    Kf  = (float*)carve((size_t)Xn * 4);
  float*    Vf  = (float*)carve((size_t)Xn * 4);
  uint16_t* Qhb = (uint16_t*)carve((size_t)Xn * 2);
  uint16_t* Khb = (uint16_t*)carve((size_t)Xn * 2);
  uint16_t* Vtb = (uint16_t*)carve((size_t)Xn * 2);
  uint16_t* Yb  = (uint16_t*)carve((size_t)Xn * 2);

  // 1) Convert X to bf16; convert + swizzle weights into B-fragment order.
  f32_to_bf16_kernel<<<(Xn / 2 + 255) / 256, 256, 0, stream>>>(x, Xb, Xn / 2);
  const int swGrid = (Dsq / 2) / 256;  // exactly 2^19 u32 outputs
  weight_swizzle_bf16_kernel<<<swGrid, 256, 0, stream>>>(Wq, Wqb);
  weight_swizzle_bf16_kernel<<<swGrid, 256, 0, stream>>>(Wk, Wkb);
  weight_swizzle_bf16_kernel<<<swGrid, 256, 0, stream>>>(Wv, Wvb);
  weight_swizzle_bf16_kernel<<<swGrid, 256, 0, stream>>>(Wo, Wob);

  // 2) Projections (Q gets 1/sqrt(D) folded in; RoPE is linear so pre-scale ok).
  dim3 ggrid(D_MOD / TBN, Mrows / TBM);
  gemm_bf16_wmma<<<ggrid, 256, 0, stream>>>(Xb, Wqb, Qf, 0.03125f);
  gemm_bf16_wmma<<<ggrid, 256, 0, stream>>>(Xb, Wkb, Kf, 1.0f);
  gemm_bf16_wmma<<<ggrid, 256, 0, stream>>>(Xb, Wvb, Vf, 1.0f);

  // 3) RoPE + head split (V transposed).
  rope_split_kernel<<<Mrows, 256, 0, stream>>>(Qf, Kf, Vf, Qhb, Khb, Vtb);

  // 4) Fused causal attention with online softmax.
  dim3 agrid(L_SEQ / 64, H_NUM, B_NUM);
  attention_wmma_kernel<<<agrid, 128, 0, stream>>>(Qhb, Khb, Vtb, Yb);

  // 5) Output projection -> fp32 result.
  gemm_bf16_wmma<<<ggrid, 256, 0, stream>>>(Yb, Wob, out, 1.0f);
}
